// Generator_10651518894593
// MI455X (gfx1250) — matrix operand order repaired and checked by reading, not compiled, not run on hardware
//
#include <hip/hip_runtime.h>
#include <hip/hip_bf16.h>

// ---------------------------------------------------------------------------
// Neural SDE (reversible Heun) for MI455X / gfx1250.
//   B=8192, STATE=128, NOISE=32, MLP=512, T=128, DATA=16
// Persistent workgroups: 128 WGs x 256 threads, each WG owns 64 batch rows
// for all 128 time points. All GEMMs via v_wmma_f32_16x16x32_bf16 (f32 accum).
// ---------------------------------------------------------------------------

typedef __attribute__((ext_vector_type(16))) __bf16 v16bf;
typedef __attribute__((ext_vector_type(8)))  __bf16 v8bf;
typedef __attribute__((ext_vector_type(8)))  float  v8f;

#define TT    128      // time points
#define BDIM  8192     // batch
#define SDIM  128      // state
#define NDIM  32       // noise
#define HDIM  512      // mlp hidden
#define DDIM  16       // readout
#define BT    64       // batch rows per workgroup
#define SA    136      // LDS stride (bf16 elems) for 128-wide activations
#define SH    520      // LDS stride (bf16 elems) for 512-wide activations

// Branch-free tanh. gfx1250 has a hardware v_tanh_f32; fall back to
// v_exp_f32 + v_rcp_f32 (no exec-mask divergence, co-executes with XDL).
__device__ __forceinline__ float fast_tanh(float x)
{
#if __has_builtin(__builtin_amdgcn_tanhf)
  return __builtin_amdgcn_tanhf(x);
#else
  float ax = __builtin_fabsf(x);
  float e  = __builtin_amdgcn_exp2f(-2.885390082f * ax);   // exp(-2|x|)
  float t  = (1.f - e) * __builtin_amdgcn_rcpf(1.f + e);
  return __builtin_copysignf(t, x);
#endif
}

// ---------------------------------------------------------------------------
// Generic 64xN = 64xK * KxN GEMM tile engine.
//  - A (activations) lives in LDS, bf16, row-major with stride SK.
//  - W lives in global memory, bf16, TRANSPOSED layout Wt[n*K + k] so that a
//    B-fragment is one contiguous 32B load per lane.
//  - 8 waves: wave&3 -> M-tile (16 rows), wave>>2 -> N half.
//  - A fragments for the whole K dimension stay resident in VGPRs.
// A/B fragments both use contiguous k = 16*half + j (not ISA table order); f32 C/D as in ISA.
// ---------------------------------------------------------------------------
template<int KTILES, typename F>
__device__ __forceinline__ void run_gemm(const __bf16* actL, int SK,
                                         const __bf16* __restrict__ Wt,
                                         int N, F post)
{
  const int lane = threadIdx.x & 31;
  const int wave = threadIdx.x >> 5;
  const int m    = wave & 3;         // M tile (rows m*16 .. m*16+15)
  const int nh   = wave >> 2;        // N half
  const int half = lane >> 4;
  const int mr   = lane & 15;
  constexpr int K = KTILES * 32;

  // A fragments: lane holds row (m*16+mr); elems j = K k0+16h+j (contiguous)
  v16bf afr[KTILES];
#pragma unroll
  for (int kt = 0; kt < KTILES; ++kt) {
    const __bf16* p = actL + (m * 16 + mr) * SK + kt * 32 + 16 * half;
    v8bf lo = *(const v8bf*)p;
    v8bf hi = *(const v8bf*)(p + 8);
    v16bf a;
#pragma unroll
    for (int j = 0; j < 8; ++j) { a[j] = lo[j]; a[j + 8] = hi[j]; }
    afr[kt] = a;
  }

  const int nBeg = nh * (N >> 1);
  const int nEnd = nBeg + (N >> 1);
  for (int n0 = nBeg; n0 < nEnd; n0 += 16) {
    // B fragment: lane = column n0+mr, elems j = K kt*32 + 16*half + j (contiguous)
    const __bf16* bp = Wt + (size_t)(n0 + mr) * K + 16 * half;
    __builtin_prefetch(bp + (size_t)16 * K, 0, 1);   // next n-tile (L2 resident stream)
    v8f acc = {};
#pragma unroll
    for (int kt = 0; kt < KTILES; ++kt) {
      v16bf b = *(const v16bf*)(bp + kt * 32);
      acc = __builtin_amdgcn_wmma_f32_16x16x32_bf16(false, afr[kt], false, b,
                                                    (short)0, acc, false, false);
    }
    // D layout: lane col = n0+mr, vgpr j row = m*16 + 8*half + j
    post(m * 16 + 8 * half, n0 + mr, acc);
  }
}

// ---------------------------------------------------------------------------
// Weight prep: f32 (K,N) row-major  ->  bf16 (N,K) transposed.
// ---------------------------------------------------------------------------
__global__ void k_transpose_bf16(const float* __restrict__ src, __bf16* __restrict__ dst,
                                 int K, int N, int rowOff, int total)
{
  int i = blockIdx.x * 256 + threadIdx.x;
  if (i >= total) return;
  int n = i / K, k = i - n * K;
  dst[i] = (__bf16)src[(size_t)(k + rowOff) * N + n];
}

__global__ void k_row0(const float* __restrict__ fW0, const float* __restrict__ gW0,
                       float* __restrict__ ftw, float* __restrict__ gtw)
{
  int i = blockIdx.x * 256 + threadIdx.x;
  if (i < HDIM) { ftw[i] = fW0[i]; gtw[i] = gW0[i]; }   // row 0 = time-feature weights
}

// ---------------------------------------------------------------------------
// Main persistent SDE kernel.
// ---------------------------------------------------------------------------
__global__ __launch_bounds__(256)
void sde_kernel(const float* __restrict__ x0, const float* __restrict__ ts,
                const float* __restrict__ dW,
                const float* __restrict__ fb0, const float* __restrict__ fb1,
                const float* __restrict__ fb2,
                const float* __restrict__ gb0, const float* __restrict__ gb1,
                const float* __restrict__ gb2,
                const float* __restrict__ rW,  const float* __restrict__ rb,
                const __bf16* __restrict__ fW0t, const __bf16* __restrict__ fW1t,
                const __bf16* __restrict__ fW2t,
                const __bf16* __restrict__ gW0t, const __bf16* __restrict__ gW1t,
                const __bf16* __restrict__ gW2t,
                const float* __restrict__ ftw, const float* __restrict__ gtw,
                float* __restrict__ out)
{
  extern __shared__ char smem[];
  float*  s_y   = (float*)smem;                 // y (with pre-added half terms)
  float*  s_z   = s_y + BT * SDIM;              // z
  float*  s_f   = s_z + BT * SDIM;              // f(t_e, X_e)
  float*  s_A   = s_f + BT * SDIM;              // A = g . dW[e]  (carried)
  float*  s_rWt = s_A + BT * SDIM;              // rW transposed (16 x 128)
  float*  s_I   = s_rWt + DDIM * SDIM;          // dW tiles, ping-pong 2 x (64x32)
  __bf16* act   = (__bf16*)(s_I + 2 * BT * NDIM);  // MLP input, bf16 (64 x 128, stride SA)
  __bf16* hidA  = act + BT * SA;                // hidden ping (64 x 512, stride SH)
  __bf16* hidB  = hidA + BT * SH;               // hidden pong

  const int tid = threadIdx.x;
  const int b0  = blockIdx.x * BT;

  // init: y0 = z0 = x0, A = 0, act = bf16(x0); load rW^T
  for (int i = tid; i < BT * SDIM; i += 256) {
    int r = i >> 7, c = i & 127;
    float v = x0[(size_t)(b0 + r) * SDIM + c];
    s_y[i] = v; s_z[i] = v; s_A[i] = 0.f;
    act[r * SA + c] = (__bf16)v;
  }
  for (int i = tid; i < DDIM * SDIM; i += 256) {
    int d = i >> 7, s = i & 127;
    s_rWt[i] = rW[s * DDIM + d];
  }
  __syncthreads();

  for (int e = 0; e < TT; ++e) {
    const float tcur = ts[e];
    const float dtp  = (e > 0)      ? tcur - ts[e - 1] : 0.f;
    const float dtn  = (e < TT - 1) ? ts[e + 1] - tcur : 0.f;
    float*       sIc = s_I + (e & 1) * (BT * NDIM);        // dW[e]
    const float* sIp = s_I + ((e + 1) & 1) * (BT * NDIM);  // dW[e-1] (prev iter's sIc)

    if (e < TT - 1) {
      for (int i = tid; i < BT * NDIM; i += 256) {
        int r = i >> 5, n = i & 31;
        sIc[i] = __builtin_nontemporal_load(
            dW + ((size_t)e * BDIM + (b0 + r)) * NDIM + n);
      }
    }
    __syncthreads();

    // ---------------- f MLP ----------------
    run_gemm<4>(act, SA, fW0t, HDIM, [&](int r0, int c, v8f acc) {
      float bb = fb0[c] + tcur * ftw[c];
#pragma unroll
      for (int j = 0; j < 8; ++j)
        hidA[(r0 + j) * SH + c] = (__bf16)fast_tanh(acc[j] + bb);
    });
    __syncthreads();

    run_gemm<16>(hidA, SH, fW1t, HDIM, [&](int r0, int c, v8f acc) {
      float bb = fb1[c];
#pragma unroll
      for (int j = 0; j < 8; ++j)
        hidB[(r0 + j) * SH + c] = (__bf16)fast_tanh(acc[j] + bb);
    });
    __syncthreads();

    run_gemm<16>(hidB, SH, fW2t, SDIM, [&](int r0, int c, v8f acc) {
      float bb = fb2[c];
#pragma unroll
      for (int j = 0; j < 8; ++j) {
        int r = r0 + j;
        float F = fast_tanh(acc[j] + bb);
        s_f[r * SDIM + c] = F;                       // unique writer per (r,c)
        if (e > 0) s_y[r * SDIM + c] += 0.5f * dtp * F;   // y_e += 0.5*dt*f1
      }
    });
    __syncthreads();

    // ---------------- g MLP ----------------
    run_gemm<4>(act, SA, gW0t, HDIM, [&](int r0, int c, v8f acc) {
      float bb = gb0[c] + tcur * gtw[c];
#pragma unroll
      for (int j = 0; j < 8; ++j)
        hidA[(r0 + j) * SH + c] = (__bf16)fast_tanh(acc[j] + bb);
    });
    __syncthreads();

    run_gemm<16>(hidA, SH, gW1t, HDIM, [&](int r0, int c, v8f acc) {
      float bb = gb1[c];
#pragma unroll
      for (int j = 0; j < 8; ++j)
        hidB[(r0 + j) * SH + c] = (__bf16)fast_tanh(acc[j] + bb);
    });
    __syncthreads();

    // Big GEMM 64x4096 with fused tanh + dual noise contraction (g never stored).
    run_gemm<16>(hidB, SH, gW2t, SDIM * NDIM, [&](int r0, int c, v8f acc) {
      float bb = gb2[c];
      int s = c >> 5, n = c & 31;   // col -> (state, noise); s uniform across lane group
      float vy[8], vA[8];
#pragma unroll
      for (int j = 0; j < 8; ++j) {
        int r = r0 + j;
        float G = fast_tanh(acc[j] + bb);
        vy[j] = (e > 0)      ? G * sIp[r * NDIM + n] : 0.f;  // g1 . I_{e-1} -> y_e
        vA[j] = (e < TT - 1) ? G * sIc[r * NDIM + n] : 0.f;  // g1 . I_e     -> A_e
      }
#pragma unroll
      for (int msk = 1; msk <= 8; msk <<= 1) {               // 16-lane column reduction
#pragma unroll
        for (int j = 0; j < 8; ++j) {
          vy[j] += __shfl_xor(vy[j], msk, 32);
          vA[j] += __shfl_xor(vA[j], msk, 32);
        }
      }
      if ((threadIdx.x & 15) == 0) {                         // lanes 0 & 16: unique writer
#pragma unroll
        for (int j = 0; j < 8; ++j) {
          int r = r0 + j;
          if (e > 0)      s_y[r * SDIM + s] += 0.5f * vy[j];
          if (e < TT - 1) s_A[r * SDIM + s] += vA[j];
        }
      }
    });
    __syncthreads();

    // ---------------- readout of finalized y_e:  out[b][e] = [t, y @ rW + rb] ----
    for (int i = tid; i < BT * (DDIM + 1); i += 256) {
      int r = i / (DDIM + 1), c = i % (DDIM + 1);
      float o;
      if (c == 0) o = tcur;
      else {
        int d = c - 1;
        float a = rb[d];
        const float* yr = s_y + r * SDIM;
        const float* wr = s_rWt + d * SDIM;
        for (int s = 0; s < SDIM; s += 4)
          a += yr[s] * wr[s] + yr[s+1] * wr[s+1] + yr[s+2] * wr[s+2] + yr[s+3] * wr[s+3];
        o = a;
      }
      __builtin_nontemporal_store(
          o, out + ((size_t)(b0 + r) * TT + e) * (DDIM + 1) + c);
    }
    __syncthreads();

    // ---------------- reversible Heun state update ----------------
    if (e < TT - 1) {
      for (int i = tid; i < BT * SDIM; i += 256) {
        int r = i >> 7, c = i & 127;
        float y = s_y[i], z = s_z[i], f = s_f[i], A = s_A[i];
        float z1 = 2.f * y - z + f * dtn + A;          // z_{e+1}
        s_z[i] = z1;
        act[r * SA + c] = (__bf16)z1;                  // next MLP input
        s_y[i] = y + 0.5f * dtn * f + 0.5f * A;        // partial y_{e+1}
        s_A[i] = 0.f;                                  // ready for next accumulation
      }
    }
    __syncthreads();
  }
}

// ---------------------------------------------------------------------------
extern "C" void kernel_launch(void* const* d_in, const int* in_sizes, int n_in,
                              void* d_out, int out_size, void* d_ws, size_t ws_size,
                              hipStream_t stream)
{
  (void)in_sizes; (void)n_in; (void)out_size; (void)ws_size;
  const float* x0  = (const float*)d_in[0];
  const float* ts  = (const float*)d_in[1];
  const float* dW  = (const float*)d_in[2];
  const float* fW0 = (const float*)d_in[3];  const float* fb0 = (const float*)d_in[4];
  const float* fW1 = (const float*)d_in[5];  const float* fb1 = (const float*)d_in[6];
  const float* fW2 = (const float*)d_in[7];  const float* fb2 = (const float*)d_in[8];
  const float* gW0 = (const float*)d_in[9];  const float* gb0 = (const float*)d_in[10];
  const float* gW1 = (const float*)d_in[11]; const float* gb1 = (const float*)d_in[12];
  const float* gW2 = (const float*)d_in[13]; const float* gb2 = (const float*)d_in[14];
  const float* rW  = (const float*)d_in[15]; const float* rb  = (const float*)d_in[16];
  float* out = (float*)d_out;

  // d_ws layout (bytes): transposed bf16 weights + layer-0 time rows (f32)
  char* ws = (char*)d_ws;
  __bf16* fW0t = (__bf16*)(ws + 0);          // 512x128
  __bf16* fW1t = (__bf16*)(ws + 131072);     // 512x512
  __bf16* fW2t = (__bf16*)(ws + 655360);     // 128x512
  __bf16* gW0t = (__bf16*)(ws + 786432);     // 512x128
  __bf16* gW1t = (__bf16*)(ws + 917504);     // 512x512
  __bf16* gW2t = (__bf16*)(ws + 1441792);    // 4096x512
  float*  ftw  = (float*)(ws + 5636096);     // 512
  float*  gtw  = (float*)(ws + 5638144);     // 512

  auto tp = [&](const float* src, __bf16* dst, int K, int N, int rowOff) {
    int total = K * N;
    k_transpose_bf16<<<(total + 255) / 256, 256, 0, stream>>>(src, dst, K, N, rowOff, total);
  };
  tp(fW0, fW0t, SDIM, HDIM, 1);         // skip time row
  tp(fW1, fW1t, HDIM, HDIM, 0);
  tp(fW2, fW2t, HDIM, SDIM, 0);
  tp(gW0, gW0t, SDIM, HDIM, 1);
  tp(gW1, gW1t, HDIM, HDIM, 0);
  tp(gW2, gW2t, HDIM, SDIM * NDIM, 0);
  k_row0<<<2, 256, 0, stream>>>(fW0, gW0, ftw, gtw);

  // Dynamic LDS: 4 f32 state arrays + rW^T + 2 dW tiles + bf16 act/hidden ping-pong
  const size_t smem =
      (size_t)(4 * BT * SDIM + DDIM * SDIM + 2 * BT * NDIM) * sizeof(float) +
      (size_t)(BT * SA + 2 * BT * SH) * sizeof(__bf16);   // = 306176 B < 320 KB

  hipFuncSetAttribute((const void*)sde_kernel,
                      hipFuncAttributeMaxDynamicSharedMemorySize, (int)smem);

  sde_kernel<<<BDIM / BT, 256, smem, stream>>>(
      x0, ts, dW, fb0, fb1, fb2, gb0, gb1, gb2, rW, rb,
      fW0t, fW1t, fW2t, gW0t, gW1t, gW2t, ftw, gtw, out);
}
